// MPNN_Bayes_75196287418585
// MI455X (gfx1250) — matrix operand order repaired and checked by reading, not compiled, not run on hardware
//
#include <hip/hip_runtime.h>
#include <hip/hip_bf16.h>
#include <math.h>

// Problem constants (match reference)
#define NN   4096
#define EE   32768
#define BB   64
#define FIN  128
#define DD   64
#define GG   10
#define LLAY 3
#define SIGMA 0.15f
#define LAMDA 0.01f
#define EPSBN 1e-5f

typedef __attribute__((ext_vector_type(16))) __bf16 v16bf;
typedef __attribute__((ext_vector_type(8)))  __bf16 v8bf;
typedef __attribute__((ext_vector_type(8)))  float  v8f;

// f32 -> bf16 round-to-nearest-even via bit manipulation (used only in the
// one-shot conversion passes, never in GEMM inner loops)
static __device__ inline __bf16 f2bf(float f) {
  unsigned u = __builtin_bit_cast(unsigned, f);
  unsigned r = (u + 0x7FFFu + ((u >> 16) & 1u)) >> 16;
  unsigned short s = (unsigned short)r;
  return __builtin_bit_cast(__bf16, s);
}

// Elementwise f32 -> bf16 (row-major A operands)
__global__ void conv_f32_bf16(const float* __restrict__ in, __bf16* __restrict__ out,
                              int n) {
  const int i = blockIdx.x * blockDim.x + threadIdx.x;
  if (i < n) out[i] = f2bf(in[i]);
}

// B operand conversion to K-contiguous layout: out[z][n][k] = B[z][k][n] (btrans=0)
// or B[z][n][k] (btrans=1, weight already stored [N,K]).
__global__ void convB_bf16(const float* __restrict__ B, int ldb, int btrans,
                           long long bstrideB, __bf16* __restrict__ out,
                           int K, int Ncols) {
  const int z = blockIdx.y;
  const int i = blockIdx.x * blockDim.x + threadIdx.x;
  if (i >= K * Ncols) return;
  const int n = i / K, k = i % K;
  const float* Bp = B + (long long)z * bstrideB;
  const float v = btrans ? Bp[(long long)n * ldb + k] : Bp[(long long)k * ldb + n];
  out[(size_t)z * K * Ncols + (size_t)n * K + k] = f2bf(v);
}

// ---------------------------------------------------------------------------
// bf16 WMMA GEMM: C[z] = op(A @ B[z] + bias + addsrc/rowscale)
// A: [M,KK] bf16 row-major.  B: [Ncols,KK] bf16 K-contiguous (batched).
// One wave per 16x16 C tile; KK/32 WMMAs, fully unrolled.
// Fragment layouts used here (wave32; contiguous k, same for A and B):
//   A 16x32 bf16: lanes 0-15 M=lane, K=0..15; lanes 16-31 same
//                 M, K=16..31  -> two 16B loads + shuffle.
//   B 32x16 bf16: lane&15 = N column; lanes 0-15 K=0..15, lanes 16-31
//                 K=16..31 -> one contiguous 32B load.
//   C/D f32: elem r -> row = 16*tm + r + 8*(lane>>4), col = 16*tn + (lane&15).
// ---------------------------------------------------------------------------
template <int KK>
__global__ void gemm_bf16_wmma(
    const __bf16* __restrict__ A,
    const __bf16* __restrict__ Bm, long long bstrideB,
    const float* __restrict__ bias,
    const float* __restrict__ addsrc, int add_ld, const float* __restrict__ addcnt,
    float* __restrict__ C, int ldc, long long bstrideC,
    int M, int Ncols, int relu)
{
  const int tilesN = Ncols >> 4;
  const int tilesM = M >> 4;
  const int wid = blockIdx.x * (blockDim.x >> 5) + (threadIdx.x >> 5);
  if (wid >= tilesM * tilesN) return;           // wave-uniform exit, EXEC stays full
  const int tm = wid / tilesN, tn = wid % tilesN;
  const __bf16* Bp = Bm + (long long)blockIdx.z * bstrideB;
  float*        Cp = C  + (long long)blockIdx.z * bstrideC;

  const int lane = threadIdx.x & 31;
  const int l15  = lane & 15;
  const int hi   = lane >> 4;
  const int bcol = tn * 16 + l15;
  const __bf16* Ar = A  + (size_t)(tm * 16 + l15) * KK + hi * 16;
  const __bf16* Br = Bp + (size_t)bcol * KK + hi * 16;

  v8f acc = {0.f, 0.f, 0.f, 0.f, 0.f, 0.f, 0.f, 0.f};
#pragma unroll
  for (int k0 = 0; k0 < KK; k0 += 32) {
    const v8bf a0 = *(const v8bf*)(Ar + k0);
    const v8bf a1 = *(const v8bf*)(Ar + k0 + 8);
    const v16bf af = __builtin_shufflevector(a0, a1, 0, 1, 2, 3, 4, 5, 6, 7,
                                             8, 9, 10, 11, 12, 13, 14, 15);
    const v16bf bf = *(const v16bf*)(Br + k0);
    acc = __builtin_amdgcn_wmma_f32_16x16x32_bf16(false, af, false, bf,
                                                  (short)0, acc, false, false);
  }

  const float bv = bias ? bias[bcol] : 0.f;
#pragma unroll
  for (int r = 0; r < 8; ++r) {
    const int row = tm * 16 + r + hi * 8;
    float v = acc[r] + bv;
    if (addsrc) {
      float scl = addcnt ? (1.f / fmaxf(addcnt[row], 1.f)) : 1.f;
      v += addsrc[(long long)row * add_ld + bcol] * scl;
    }
    if (relu) v = fmaxf(v, 0.f);
    Cp[(long long)row * ldc + bcol] = v;
  }
}

// msg[e,:] = h1[e,:] @ P[src(e)] + Q[src(e),:]  then atomic scatter into aggsum[dst]
// One wave per edge; lane covers columns {lane, lane+32}. P rows hit L2 (64 MB < 192 MB).
__global__ void edge_msg(const float* __restrict__ h1, const float* __restrict__ P,
                         const float* __restrict__ Q, const int* __restrict__ src,
                         const int* __restrict__ dst, float* __restrict__ aggsum, int E)
{
  const int wid  = blockIdx.x * (blockDim.x >> 5) + (threadIdx.x >> 5);
  const int nw   = gridDim.x * (blockDim.x >> 5);
  const int lane = threadIdx.x & 31;
  for (int e = wid; e < E; e += nw) {
    const int s = src[e], d = dst[e];
    const float* Pr = P + (size_t)s * (DD * DD);
    const float* hr = h1 + (size_t)e * DD;
    float a0 = Q[s * DD + lane];
    float a1 = Q[s * DD + 32 + lane];
    for (int k = 0; k < DD; ++k) {
      const float hk = hr[k];                 // wave-uniform broadcast
      a0 = fmaf(hk, Pr[k * DD + lane], a0);
      a1 = fmaf(hk, Pr[k * DD + 32 + lane], a1);
    }
    atomicAdd(&aggsum[(size_t)d * DD + lane], a0);
    atomicAdd(&aggsum[(size_t)d * DD + 32 + lane], a1);
  }
}

__global__ void count_seg(const int* __restrict__ idx, float* __restrict__ cnt, int n) {
  const int i = blockIdx.x * blockDim.x + threadIdx.x;
  if (i < n) atomicAdd(&cnt[idx[i]], 1.0f);
}

// s = softmax(m @ gn_lin)  per node (G=10 logits, K=64)
__global__ void dgn_softmax(const float* __restrict__ m, const float* __restrict__ lin,
                            float* __restrict__ s) {
  const int n = blockIdx.x * blockDim.x + threadIdx.x;
  if (n >= NN) return;
  float logit[GG];
#pragma unroll
  for (int g = 0; g < GG; ++g) logit[g] = 0.f;
  const float* mr = m + (size_t)n * DD;
  for (int f = 0; f < DD; ++f) {
    const float mv = mr[f];
#pragma unroll
    for (int g = 0; g < GG; ++g) logit[g] = fmaf(mv, lin[f * GG + g], logit[g]);
  }
  float mx = logit[0];
#pragma unroll
  for (int g = 1; g < GG; ++g) mx = fmaxf(mx, logit[g]);
  float sum = 0.f;
#pragma unroll
  for (int g = 0; g < GG; ++g) { logit[g] = __expf(logit[g] - mx); sum += logit[g]; }
  const float inv = 1.f / sum;
#pragma unroll
  for (int g = 0; g < GG; ++g) s[(size_t)n * GG + g] = logit[g] * inv;
}

// Column sums of t=s[:,g]*m[:,d] and t^2 over nodes (64 nodes/block staged in LDS)
__global__ void dgn_stats(const float* __restrict__ m, const float* __restrict__ s,
                          float* __restrict__ sum1, float* __restrict__ sum2) {
  __shared__ float sm[64 * DD];
  __shared__ float ss[64 * GG];
  const int t = threadIdx.x;             // blockDim = 640
  const int base = blockIdx.x * 64;
  for (int i = t; i < 64 * DD; i += blockDim.x) sm[i] = m[(size_t)base * DD + i];
  for (int i = t; i < 64 * GG; i += blockDim.x) ss[i] = s[(size_t)base * GG + i];
  __syncthreads();
  const int g = t / DD, d = t % DD;
  float a1 = 0.f, a2 = 0.f;
  for (int j = 0; j < 64; ++j) {
    const float v = ss[j * GG + g] * sm[j * DD + d];
    a1 += v; a2 = fmaf(v, v, a2);
  }
  atomicAdd(&sum1[t], a1);
  atomicAdd(&sum2[t], a2);
}

// a[g,d] = gamma/sqrt(var+eps); c[d] = sum_g (beta - mu*a)
__global__ void dgn_finalize(const float* __restrict__ sum1, const float* __restrict__ sum2,
                             const float* __restrict__ gamma, const float* __restrict__ beta,
                             float* __restrict__ a, float* __restrict__ c) {
  const int t = threadIdx.x;             // single block, 640 threads
  if (t < DD) c[t] = 0.f;
  __syncthreads();
  const float mu  = sum1[t] * (1.0f / NN);
  const float var = sum2[t] * (1.0f / NN) - mu * mu;
  const float av  = gamma[t] * rsqrtf(var + EPSBN);
  a[t] = av;
  atomicAdd(&c[t % DD], beta[t] - mu * av);
}

// m = relu( m + lamda*( m*(s@a) + c ) )
__global__ void dgn_apply_relu(float* __restrict__ m, const float* __restrict__ s,
                               const float* __restrict__ a, const float* __restrict__ c) {
  const int i = blockIdx.x * blockDim.x + threadIdx.x;
  if (i >= NN * DD) return;
  const int n = i / DD, d = i % DD;
  float sa = 0.f;
#pragma unroll
  for (int g = 0; g < GG; ++g) sa = fmaf(s[(size_t)n * GG + g], a[g * DD + d], sa);
  const float mv = m[i];
  const float r = mv + LAMDA * fmaf(mv, sa, c[d]);
  m[i] = fmaxf(r, 0.f);
}

// GRU gates (torch order r,z,n) + residual out += h_new, h = h_new
__global__ void gru_residual(const float* __restrict__ gi, const float* __restrict__ gh,
                             float* __restrict__ h, float* __restrict__ out) {
  const int i = blockIdx.x * blockDim.x + threadIdx.x;
  if (i >= NN * DD) return;
  const int n = i / DD, d = i % DD;
  const float* gir = gi + (size_t)n * 3 * DD;
  const float* ghr = gh + (size_t)n * 3 * DD;
  const float r = 1.f / (1.f + __expf(-(gir[d] + ghr[d])));
  const float z = 1.f / (1.f + __expf(-(gir[DD + d] + ghr[DD + d])));
  const float nn = tanhf(gir[2 * DD + d] + r * ghr[2 * DD + d]);
  const float hnew = (1.f - z) * nn + z * h[i];
  h[i] = hnew;
  out[i] = out[i] + hnew;
}

__global__ void pool_accum(const float* __restrict__ out, const int* __restrict__ batch,
                           float* __restrict__ psum) {
  const int i = blockIdx.x * blockDim.x + threadIdx.x;
  if (i >= NN * DD) return;
  const int n = i / DD, d = i % DD;
  atomicAdd(&psum[(size_t)batch[n] * DD + d], out[i]);
}

__global__ void pool_final(const float* __restrict__ psum, const float* __restrict__ cnt,
                           float* __restrict__ pooled) {
  const int i = blockIdx.x * blockDim.x + threadIdx.x;
  if (i >= BB * DD) return;
  pooled[i] = psum[i] / fmaxf(cnt[i / DD], 1.f);
}

__global__ void final_out(const float* __restrict__ p, const float* __restrict__ oW,
                          const float* __restrict__ ob, float* __restrict__ y) {
  const int b = threadIdx.x;
  if (b >= BB) return;
  float acc = ob[0];
  for (int d = 0; d < DD; ++d) acc = fmaf(p[b * DD + d], oW[d], acc);
  y[b] = acc;
}

// tkl = 0.5/sig^2 * sum(w^2) + count*log(sig*sqrt(2*pi)) over all Bayes params
__global__ void kld_kernel(const float* w0, int n0, const float* w1, int n1,
                           const float* w2, int n2, const float* w3, int n3,
                           const float* w4, int n4, const float* w5, int n5,
                           float* __restrict__ outv) {
  __shared__ float red[256];
  const int t = threadIdx.x;
  const float* wp[6] = {w0, w1, w2, w3, w4, w5};
  const int    np[6] = {n0, n1, n2, n3, n4, n5};
  float acc = 0.f;
  for (int a = 0; a < 6; ++a)
    for (int i = t; i < np[a]; i += 256) { const float v = wp[a][i]; acc = fmaf(v, v, acc); }
  red[t] = acc;
  __syncthreads();
  for (int o = 128; o > 0; o >>= 1) { if (t < o) red[t] += red[t + o]; __syncthreads(); }
  if (t == 0) {
    const float total = (float)(n0 + n1 + n2 + n3 + n4 + n5);
    const float logc = logf(SIGMA * sqrtf(2.0f * 3.14159265358979323846f));
    outv[0] = 0.5f * red[0] / (SIGMA * SIGMA) + total * logc;
  }
}

// ---------------------------------------------------------------------------
static inline void gemm_launch(hipStream_t stream,
    const __bf16* A, const __bf16* B, long long bsB,
    const float* bias, const float* addsrc, int add_ld, const float* addcnt,
    float* C, int ldc, long long bsC, int M, int Ncols, int K, int relu, int nbatch)
{
  const int tiles = (M >> 4) * (Ncols >> 4);
  dim3 grid((tiles + 7) / 8, 1, nbatch);       // 8 waves (256 thr) per block
  if (K == 128)
    gemm_bf16_wmma<128><<<grid, 256, 0, stream>>>(A, B, bsB, bias, addsrc, add_ld,
                                                  addcnt, C, ldc, bsC, M, Ncols, relu);
  else
    gemm_bf16_wmma<64><<<grid, 256, 0, stream>>>(A, B, bsB, bias, addsrc, add_ld,
                                                 addcnt, C, ldc, bsC, M, Ncols, relu);
}

static inline void convA(hipStream_t s, const float* in, __bf16* out, int n) {
  conv_f32_bf16<<<(n + 255) / 256, 256, 0, s>>>(in, out, n);
}
static inline void convB(hipStream_t s, const float* B, int ldb, int btrans,
                         long long bsB, __bf16* out, int K, int N, int nbatch) {
  dim3 grid((K * N + 255) / 256, nbatch);
  convB_bf16<<<grid, 256, 0, s>>>(B, ldb, btrans, bsB, out, K, N);
}

extern "C" void kernel_launch(void* const* d_in, const int* in_sizes, int n_in,
                              void* d_out, int out_size, void* d_ws, size_t ws_size,
                              hipStream_t stream) {
  (void)in_sizes; (void)n_in; (void)out_size; (void)ws_size;
  const float* x         = (const float*)d_in[0];
  const float* edge_attr = (const float*)d_in[1];
  const float* pre_W     = (const float*)d_in[2];
  const float* pre_b     = (const float*)d_in[3];
  const float* edge_w1   = (const float*)d_in[4];
  const float* edge_b1   = (const float*)d_in[5];
  const float* edge_w2   = (const float*)d_in[6];
  const float* edge_b2   = (const float*)d_in[7];
  const float* root_w    = (const float*)d_in[8];
  const float* root_b    = (const float*)d_in[9];
  const float* gru_wih   = (const float*)d_in[10];
  const float* gru_whh   = (const float*)d_in[11];
  const float* gru_bih   = (const float*)d_in[12];
  const float* gru_bhh   = (const float*)d_in[13];
  const float* gn_lin    = (const float*)d_in[14];
  const float* gn_gamma  = (const float*)d_in[15];
  const float* gn_beta   = (const float*)d_in[16];
  const float* post_W    = (const float*)d_in[17];
  const float* post_b    = (const float*)d_in[18];
  const float* out_W     = (const float*)d_in[19];
  const float* out_b     = (const float*)d_in[20];
  const int*   eidx      = (const int*)d_in[21];
  const int*   batch     = (const int*)d_in[22];
  const int* srcp = eidx;
  const int* dstp = eidx + EE;
  float* yout = (float*)d_out;

  // ---- workspace carve-out: f32 region (~87 MB) then bf16 region (~9 MB) ----
  float* w = (float*)d_ws;
  size_t off = 0;
  auto alloc = [&](size_t n) { float* p = w + off; off += n; return p; };
  float* out   = alloc((size_t)NN * DD);
  float* h     = alloc((size_t)NN * DD);
  float* h1    = alloc((size_t)EE * DD);
  float* P     = alloc((size_t)NN * DD * DD);   // 64 MB: L2-resident
  float* Q     = alloc((size_t)NN * DD);
  float* agg   = alloc((size_t)NN * DD);
  float* deg   = alloc((size_t)NN);
  float* mbuf  = alloc((size_t)NN * DD);
  float* sfm   = alloc((size_t)NN * GG);
  float* sum1  = alloc((size_t)GG * DD);
  float* sum2  = alloc((size_t)GG * DD);
  float* abuf  = alloc((size_t)GG * DD);
  float* cbuf  = alloc((size_t)DD);
  float* gi    = alloc((size_t)NN * 3 * DD);
  float* gh    = alloc((size_t)NN * 3 * DD);
  float* psum  = alloc((size_t)BB * DD);
  float* bcnt  = alloc((size_t)BB);
  float* pooled= alloc((size_t)BB * DD);
  float* pbuf  = alloc((size_t)BB * DD);
  off = (off + 15) & ~(size_t)15;               // 64B-align bf16 region
  __bf16* bw = (__bf16*)(w + off);
  size_t boff = 0;
  auto balloc = [&](size_t n) { __bf16* p = bw + boff; boff += (n + 15) & ~(size_t)15; return p; };
  __bf16* xbf     = balloc((size_t)NN * FIN);
  __bf16* eabf    = balloc((size_t)EE * DD);
  __bf16* outbf   = balloc((size_t)NN * DD);
  __bf16* hbf     = balloc((size_t)NN * DD);
  __bf16* mbf     = balloc((size_t)NN * DD);
  __bf16* pooledbf= balloc((size_t)BB * DD);
  __bf16* preWbf  = balloc((size_t)FIN * DD);
  __bf16* postWbf = balloc((size_t)DD * DD);
  __bf16* ew1bf   = balloc((size_t)LLAY * DD * DD);
  __bf16* eb2bf   = balloc((size_t)LLAY * DD * DD);
  __bf16* ew2bf   = balloc((size_t)LLAY * DD * DD * DD);
  __bf16* rwbf    = balloc((size_t)LLAY * DD * DD);
  __bf16* wihbf   = balloc((size_t)LLAY * 3 * DD * DD);
  __bf16* whhbf   = balloc((size_t)LLAY * 3 * DD * DD);

  // ---- one-shot operand conversions (weights + static activations) ----
  convA(stream, x, xbf, NN * FIN);
  convA(stream, edge_attr, eabf, EE * DD);
  convB(stream, pre_W, DD, 0, 0, preWbf, FIN, DD, 1);
  convB(stream, post_W, DD, 0, 0, postWbf, DD, DD, 1);
  for (int i = 0; i < LLAY; ++i) {
    convB(stream, edge_w1 + (size_t)i * DD * DD, DD, 0, 0,
          ew1bf + (size_t)i * DD * DD, DD, DD, 1);
    convB(stream, edge_b2 + (size_t)i * DD * DD, DD, 0, 0,
          eb2bf + (size_t)i * DD * DD, DD, DD, 1);
    convB(stream, edge_w2 + (size_t)i * DD * DD * DD, DD, 0, (long long)DD * DD,
          ew2bf + (size_t)i * DD * DD * DD, DD, DD, DD);   // 64 slices
    convB(stream, root_w + (size_t)i * DD * DD, DD, 0, 0,
          rwbf + (size_t)i * DD * DD, DD, DD, 1);
    convB(stream, gru_wih + (size_t)i * 3 * DD * DD, DD, 1, 0,
          wihbf + (size_t)i * 3 * DD * DD, DD, 3 * DD, 1); // [192,64] already [N,K]
    convB(stream, gru_whh + (size_t)i * 3 * DD * DD, DD, 1, 0,
          whhbf + (size_t)i * 3 * DD * DD, DD, 3 * DD, 1);
  }

  // pre FC: out = relu(x @ pre_W + pre_b); h = out
  gemm_launch(stream, xbf, preWbf, 0, pre_b, nullptr, 0, nullptr,
              out, DD, 0, NN, DD, FIN, 1, 1);
  hipMemcpyAsync(h, out, (size_t)NN * DD * sizeof(float),
                 hipMemcpyDeviceToDevice, stream);

  // in-degree counts (scatter-mean denominator)
  hipMemsetAsync(deg, 0, NN * sizeof(float), stream);
  count_seg<<<(EE + 255) / 256, 256, 0, stream>>>(dstp, deg, EE);

  for (int i = 0; i < LLAY; ++i) {
    const float* eb1 = edge_b1 + (size_t)i * DD;
    const float* rb  = root_b  + (size_t)i * DD;
    const float* bih = gru_bih + (size_t)i * 3 * DD;
    const float* bhh = gru_bhh + (size_t)i * 3 * DD;
    const float* lin = gn_lin  + (size_t)i * DD * GG;
    const float* gam = gn_gamma+ (size_t)i * GG * DD;
    const float* bet = gn_beta + (size_t)i * GG * DD;

    convA(stream, out, outbf, NN * DD);
    convA(stream, h,   hbf,   NN * DD);

    // h1 = relu(edge_attr @ ew1 + eb1)            [32768,64]@[64,64]
    gemm_launch(stream, eabf, ew1bf + (size_t)i * DD * DD, 0, eb1,
                nullptr, 0, nullptr, h1, DD, 0, EE, DD, DD, 1, 1);
    // Q = out @ b2(as [64,64])                    [4096,64]@[64,64]
    gemm_launch(stream, outbf, eb2bf + (size_t)i * DD * DD, 0, nullptr,
                nullptr, 0, nullptr, Q, DD, 0, NN, DD, DD, 0, 1);
    // P[:,k,:] = out @ w2[k]  (batched over k)    64 x [4096,64]@[64,64]
    gemm_launch(stream, outbf, ew2bf + (size_t)i * DD * DD * DD,
                (long long)DD * DD, nullptr, nullptr, 0, nullptr,
                P, DD * DD, (long long)DD, NN, DD, DD, 0, DD);
    // agg = scatter-add_e ( h1[e] @ P[src] + Q[src] ) into dst
    hipMemsetAsync(agg, 0, (size_t)NN * DD * sizeof(float), stream);
    edge_msg<<<EE / 8, 256, 0, stream>>>(h1, P, Q, srcp, dstp, agg, EE);
    // m = out @ root_w + root_b + agg/deg
    gemm_launch(stream, outbf, rwbf + (size_t)i * DD * DD, 0, rb, agg, DD, deg,
                mbuf, DD, 0, NN, DD, DD, 0, 1);
    // DiffGroupNorm + relu
    dgn_softmax<<<(NN + 255) / 256, 256, 0, stream>>>(mbuf, lin, sfm);
    hipMemsetAsync(sum1, 0, GG * DD * sizeof(float), stream);
    hipMemsetAsync(sum2, 0, GG * DD * sizeof(float), stream);
    dgn_stats<<<NN / 64, GG * DD, 0, stream>>>(mbuf, sfm, sum1, sum2);
    dgn_finalize<<<1, GG * DD, 0, stream>>>(sum1, sum2, gam, bet, abuf, cbuf);
    dgn_apply_relu<<<(NN * DD + 255) / 256, 256, 0, stream>>>(mbuf, sfm, abuf, cbuf);
    // GRU: gi = m @ wih^T + bih ; gh = h @ whh^T + bhh   [4096,64]@[64,192]
    convA(stream, mbuf, mbf, NN * DD);
    gemm_launch(stream, mbf, wihbf + (size_t)i * 3 * DD * DD, 0, bih,
                nullptr, 0, nullptr, gi, 3 * DD, 0, NN, 3 * DD, DD, 0, 1);
    gemm_launch(stream, hbf, whhbf + (size_t)i * 3 * DD * DD, 0, bhh,
                nullptr, 0, nullptr, gh, 3 * DD, 0, NN, 3 * DD, DD, 0, 1);
    gru_residual<<<(NN * DD + 255) / 256, 256, 0, stream>>>(gi, gh, h, out);
  }

  // global mean pool + post MLP
  hipMemsetAsync(psum, 0, (size_t)BB * DD * sizeof(float), stream);
  hipMemsetAsync(bcnt, 0, BB * sizeof(float), stream);
  count_seg<<<(NN + 255) / 256, 256, 0, stream>>>(batch, bcnt, NN);
  pool_accum<<<(NN * DD + 255) / 256, 256, 0, stream>>>(out, batch, psum);
  pool_final<<<(BB * DD + 255) / 256, 256, 0, stream>>>(psum, bcnt, pooled);
  convA(stream, pooled, pooledbf, BB * DD);
  // p = relu(pooled @ post_W + post_b)            [64,64]@[64,64]
  gemm_launch(stream, pooledbf, postWbf, 0, post_b, nullptr, 0, nullptr,
              pbuf, DD, 0, BB, DD, DD, 1, 1);
  // y = p @ out_W + out_b
  final_out<<<1, BB, 0, stream>>>(pbuf, out_W, out_b, yout);
  // KL term -> d_out[64]
  kld_kernel<<<1, 256, 0, stream>>>(pre_W, FIN * DD, pre_b, DD,
                                    post_W, DD * DD, post_b, DD,
                                    out_W, DD, out_b, 1, yout + BB);
}
